// SwarmNet_44822278701437
// MI455X (gfx1250) — matrix operand order repaired and checked by reading, not compiled, not run on hardware
//
#include <hip/hip_runtime.h>

// ---------------------------------------------------------------------------
// SwarmNet-style rollout for MI455X (gfx1250, wave32, WMMA).
// Dominant cost: A_e[4096x4096] @ cond_b[4096x64] (E*B=6 pairs, 10 steps).
// A is converted to f16 once per launch; big GEMM + encoder/decoder GEMMs all
// run on v_wmma_f32_16x16x32_f16 with f32 accumulation. The big GEMM is
// software-pipelined with a ping-pong operand double buffer (unroll-by-2) so
// WMMAs overlap loads without register-rotation copies.
// ---------------------------------------------------------------------------

typedef __attribute__((ext_vector_type(16))) _Float16 v16h;
typedef __attribute__((ext_vector_type(8)))  _Float16 v8h;
typedef __attribute__((ext_vector_type(8)))  float    v8f;

#define NN 4096
#define BB 2
#define TT 5
#define DD 4
#define FF 64
#define HH 64
#define EE 3
#define PP 10

union V16 { v16h v; v8h h[2]; };

static __device__ __forceinline__ v8f wmma_f16(v16h a, v16h b, v8f c) {
  return __builtin_amdgcn_wmma_f32_16x16x32_f16(false, a, false, b, (short)0, c,
                                                false, false);
}

// ------------------------------- prep kernels ------------------------------

__global__ void prep_adj_kernel(const float* __restrict__ adjs,
                                _Float16* __restrict__ A16, int n) {
  int id = blockIdx.x * 256 + threadIdx.x;
  if (id < n) A16[id] = (_Float16)adjs[id];
}

__global__ void prep_mask_kernel(const float* __restrict__ adjs,
                                 float* __restrict__ mask) {
  int n = blockIdx.x * 256 + threadIdx.x;  // n < NN, coalesced column reads
  float s = 0.f;
  for (int em = 0; em < EE * NN; ++em) s += adjs[(size_t)em * NN + n];
  mask[n] = (s > 0.f) ? 1.f : 0.f;
}

__global__ void prep_w_kernel(const float* __restrict__ enc_w1,
                              const float* __restrict__ enc_w2,
                              const float* __restrict__ dec_w,
                              _Float16* __restrict__ w1T,
                              _Float16* __restrict__ w2T,
                              _Float16* __restrict__ decT) {
  int id = blockIdx.x * 256 + threadIdx.x;
  const int EHF = EE * HH * FF;
  if (id < EHF) {
    int e = id / (HH * FF);
    int r = id % (HH * FF);
    int h = r / FF, f = r % FF;
    w1T[id] = (_Float16)enc_w1[((size_t)e * FF + f) * HH + h];
    w2T[id] = (_Float16)enc_w2[((size_t)e * FF + f) * HH + h];
  } else if (id < EHF + HH * (FF + HH)) {
    int r = id - EHF;
    int h = r / (FF + HH), k = r % (FF + HH);
    decT[r] = (_Float16)dec_w[(size_t)k * HH + h];
  }
}

__global__ void prep_win_kernel(const float* __restrict__ ts,
                                float* __restrict__ win) {
  int id = blockIdx.x * 256 + threadIdx.x;  // enumerates [B][N][T][D]
  if (id >= BB * NN * TT * DD) return;
  int d = id & 3;
  int t = (id >> 2) % TT;
  int n = (id / (DD * TT)) & (NN - 1);
  int b = id / (DD * TT * NN);
  win[id] = ts[(((size_t)b * TT + t) * NN + n) * DD + d];
}

// ------------------------------ per-step: cond -----------------------------

__global__ void cond_kernel(const float* __restrict__ win,
                            const float* __restrict__ conv_w,
                            const float* __restrict__ conv_b,
                            _Float16* __restrict__ cond16,
                            _Float16* __restrict__ condT16) {
  int id = blockIdx.x * 256 + threadIdx.x;  // B*N*F threads
  int f = id & (FF - 1);
  int n = (id >> 6) & (NN - 1);
  int b = id >> 18;
  const float* w = win + ((size_t)b * NN + n) * TT * DD;
  float s = conv_b[f];
#pragma unroll
  for (int td = 0; td < TT * DD; ++td) s += w[td] * conv_w[td * FF + f];
  cond16[((size_t)b * NN + n) * FF + f] = (_Float16)s;
  condT16[((size_t)b * FF + f) * NN + n] = (_Float16)s;
}

// -------------------- per-step: agg = A_e @ cond_b (WMMA) ------------------
// One wave computes a 32(node) x 64(F) block: 2 M-tiles x 4 F-tiles.
// K = 4096 in chunks of 32; the loop is unrolled by 2 with ping-pong operand
// buffers so loads for one chunk overlap the 8 WMMAs of the other.

struct Ops {
  V16 a0, a1;
  v16h b0, b1, b2, b3;
};

static __device__ __forceinline__ void load_ops(Ops& o,
                                                const _Float16* __restrict__ a0,
                                                const _Float16* __restrict__ a1,
                                                const _Float16* __restrict__ bp,
                                                int k) {
  o.a0.h[0] = *(const v8h*)(a0 + k);
  o.a0.h[1] = *(const v8h*)(a0 + k + 8);
  o.a1.h[0] = *(const v8h*)(a1 + k);
  o.a1.h[1] = *(const v8h*)(a1 + k + 8);
  o.b0 = *(const v16h*)(bp + k);
  o.b1 = *(const v16h*)(bp + k + 16 * NN);
  o.b2 = *(const v16h*)(bp + k + 32 * NN);
  o.b3 = *(const v16h*)(bp + k + 48 * NN);
}

static __device__ __forceinline__ void do_wmma8(v8f acc[8], const Ops& o) {
  acc[0] = wmma_f16(o.a0.v, o.b0, acc[0]);
  acc[1] = wmma_f16(o.a0.v, o.b1, acc[1]);
  acc[2] = wmma_f16(o.a0.v, o.b2, acc[2]);
  acc[3] = wmma_f16(o.a0.v, o.b3, acc[3]);
  acc[4] = wmma_f16(o.a1.v, o.b0, acc[4]);
  acc[5] = wmma_f16(o.a1.v, o.b1, acc[5]);
  acc[6] = wmma_f16(o.a1.v, o.b2, acc[6]);
  acc[7] = wmma_f16(o.a1.v, o.b3, acc[7]);
}

__global__ __launch_bounds__(256) void agg_gemm_kernel(
    const _Float16* __restrict__ A16, const _Float16* __restrict__ condT,
    _Float16* __restrict__ agg16) {
  const int lane = threadIdx.x & 31;
  const int warp = threadIdx.x >> 5;
  const int mtile = (blockIdx.x * 8 + warp) * 2;  // 0,2,..,254
  const int eb = blockIdx.y;                      // e*2 + b
  const int e = eb >> 1, b = eb & 1;
  const int lh = lane & 15, hi = lane >> 4;

  // A operand (16x32 f16): lane holds row M=lh, K = hi*16 .. +15 (contiguous)
  const _Float16* aptr0 =
      A16 + (size_t)e * NN * NN + (size_t)(mtile * 16 + lh) * NN + hi * 16;
  const _Float16* aptr1 = aptr0 + (size_t)16 * NN;
  // B operand (32x16 f16): lane holds column lh, K = hi*16 .. +15 (contiguous)
  const _Float16* bptr =
      condT + (size_t)b * FF * NN + (size_t)lh * NN + hi * 16;

  v8f acc[8] = {v8f{}, v8f{}, v8f{}, v8f{}, v8f{}, v8f{}, v8f{}, v8f{}};

  Ops buf0, buf1;
  load_ops(buf0, aptr0, aptr1, bptr, 0);
  int k = 0;
  for (; k < NN - 64; k += 64) {
    load_ops(buf1, aptr0, aptr1, bptr, k + 32);
    do_wmma8(acc, buf0);
    load_ops(buf0, aptr0, aptr1, bptr, k + 64);
    do_wmma8(acc, buf1);
  }
  // k == NN-64: last pair
  load_ops(buf1, aptr0, aptr1, bptr, k + 32);
  do_wmma8(acc, buf0);
  do_wmma8(acc, buf1);

  _Float16* og = agg16 + (size_t)eb * NN * FF;
#pragma unroll
  for (int mt = 0; mt < 2; ++mt) {
#pragma unroll
    for (int v = 0; v < 8; ++v) {
      int m = (mtile + mt) * 16 + v + hi * 8;  // C-layout row
      og[(size_t)m * FF + 0 + lh] = (_Float16)acc[mt * 4 + 0][v];
      og[(size_t)m * FF + 16 + lh] = (_Float16)acc[mt * 4 + 1][v];
      og[(size_t)m * FF + 32 + lh] = (_Float16)acc[mt * 4 + 2][v];
      og[(size_t)m * FF + 48 + lh] = (_Float16)acc[mt * 4 + 3][v];
    }
  }
}

// --------- per-step: fused encoder + decoder + head + window shift ---------
// One wave handles 16 nodes of one batch. LDS used to re-shape C-layout
// results into A-operand layout for the K=128 decoder GEMM.

__global__ __launch_bounds__(256) void encdec_kernel(
    const _Float16* __restrict__ agg16, const _Float16* __restrict__ cond16,
    const _Float16* __restrict__ w1T, const _Float16* __restrict__ w2T,
    const _Float16* __restrict__ decT, const float* __restrict__ enc_b,
    const float* __restrict__ dec_b, const float* __restrict__ out_w,
    const float* __restrict__ out_b, const float* __restrict__ mask,
    float* __restrict__ win, float* __restrict__ preds, int step) {
  __shared__ __align__(32) _Float16 smem[8 * 2048];
  const int lane = threadIdx.x & 31;
  const int warp = threadIdx.x >> 5;
  const int gt = blockIdx.x * 8 + warp;  // 0 .. B*N/16-1
  const int b = gt / (NN / 16);
  const int ntile = gt % (NN / 16);
  const int lh = lane & 15, hi = lane >> 4;
  _Float16* enc_l = smem + warp * 2048;  // 16x64 encoded tile
  _Float16* h_l = enc_l + 1024;          // 16x64 hidden tile

  const int row = ntile * 16 + lh;

  // cond A-operands for K chunks 0,1 (reused by encoder w2 term and decoder)
  const _Float16* cptr = cond16 + ((size_t)b * NN + row) * FF + hi * 16;
  V16 condA0, condA1;
  condA0.h[0] = *(const v8h*)(cptr);
  condA0.h[1] = *(const v8h*)(cptr + 8);
  condA1.h[0] = *(const v8h*)(cptr + 32);
  condA1.h[1] = *(const v8h*)(cptr + 40);

  v8f esum[4] = {v8f{}, v8f{}, v8f{}, v8f{}};
  for (int e = 0; e < EE; ++e) {
    const _Float16* gptr =
        agg16 + (((size_t)e * BB + b) * NN + row) * FF + hi * 16;
    V16 aggA0, aggA1;
    aggA0.h[0] = *(const v8h*)(gptr);
    aggA0.h[1] = *(const v8h*)(gptr + 8);
    aggA1.h[0] = *(const v8h*)(gptr + 32);
    aggA1.h[1] = *(const v8h*)(gptr + 40);
#pragma unroll
    for (int ht = 0; ht < 4; ++ht) {
      float bias = enc_b[e * HH + ht * 16 + lh];
      v8f acc = {bias, bias, bias, bias, bias, bias, bias, bias};
      const _Float16* w1p = w1T + ((size_t)e * HH + ht * 16 + lh) * FF + hi * 16;
      const _Float16* w2p = w2T + ((size_t)e * HH + ht * 16 + lh) * FF + hi * 16;
      acc = wmma_f16(aggA0.v, *(const v16h*)(w1p), acc);
      acc = wmma_f16(aggA1.v, *(const v16h*)(w1p + 32), acc);
      acc = wmma_f16(condA0.v, *(const v16h*)(w2p), acc);
      acc = wmma_f16(condA1.v, *(const v16h*)(w2p + 32), acc);
#pragma unroll
      for (int i = 0; i < 8; ++i) esum[ht][i] += tanhf(acc[i]);
    }
  }

  // encoded = tanh(sum_e per) * mask ; scatter C-layout -> LDS row-major
  float mk[8];
#pragma unroll
  for (int v = 0; v < 8; ++v) mk[v] = mask[ntile * 16 + v + hi * 8];
#pragma unroll
  for (int ht = 0; ht < 4; ++ht) {
#pragma unroll
    for (int v = 0; v < 8; ++v) {
      int m = v + hi * 8;
      enc_l[m * FF + ht * 16 + lh] = (_Float16)(tanhf(esum[ht][v]) * mk[v]);
    }
  }
  asm volatile("s_wait_dscnt 0" ::: "memory");

  // decoder: h = relu([cond | encoded] @ dec_w + dec_b), K = 128
  V16 encA0, encA1;
  {
    const _Float16* ep = enc_l + lh * FF + hi * 16;
    encA0.h[0] = *(const v8h*)(ep);
    encA0.h[1] = *(const v8h*)(ep + 8);
    encA1.h[0] = *(const v8h*)(ep + 32);
    encA1.h[1] = *(const v8h*)(ep + 40);
  }
#pragma unroll
  for (int ht = 0; ht < 4; ++ht) {
    float bias = dec_b[ht * 16 + lh];
    v8f acc = {bias, bias, bias, bias, bias, bias, bias, bias};
    const _Float16* dp = decT + (size_t)(ht * 16 + lh) * (FF + HH) + hi * 16;
    acc = wmma_f16(condA0.v, *(const v16h*)(dp), acc);
    acc = wmma_f16(condA1.v, *(const v16h*)(dp + 32), acc);
    acc = wmma_f16(encA0.v, *(const v16h*)(dp + 64), acc);
    acc = wmma_f16(encA1.v, *(const v16h*)(dp + 96), acc);
#pragma unroll
    for (int v = 0; v < 8; ++v)
      h_l[(v + hi * 8) * HH + ht * 16 + lh] = (_Float16)fmaxf(acc[v], 0.f);
  }
  asm volatile("s_wait_dscnt 0" ::: "memory");

  // head: nxt = win_last + tanh(h @ out_w + out_b); window shift; preds
#pragma unroll
  for (int j = 0; j < 2; ++j) {
    int idx = lane * 2 + j;  // 0..63 -> (node m, dim d)
    int m = idx >> 2, d = idx & 3;
    const _Float16* hrow = h_l + m * HH;
    float s = 0.f;
#pragma unroll
    for (int kk = 0; kk < 8; ++kk) {
      v8h hv = *(const v8h*)(hrow + kk * 8);
#pragma unroll
      for (int q = 0; q < 8; ++q) s += (float)hv[q] * out_w[(kk * 8 + q) * DD + d];
    }
    float val = tanhf(s + out_b[d]);
    int gn = ntile * 16 + m;
    size_t wbase = (((size_t)b * NN + gn) * TT) * DD + d;
    float nxt = win[wbase + (TT - 1) * DD] + val;
#pragma unroll
    for (int t = 0; t < TT - 1; ++t) win[wbase + t * DD] = win[wbase + (t + 1) * DD];
    win[wbase + (TT - 1) * DD] = nxt;
    preds[((((size_t)b * PP) + step) * NN + gn) * DD + d] = nxt;
  }
}

// --------------------------------- launch ----------------------------------

extern "C" void kernel_launch(void* const* d_in, const int* in_sizes, int n_in,
                              void* d_out, int out_size, void* d_ws,
                              size_t ws_size, hipStream_t stream) {
  const float* time_segs = (const float*)d_in[0];
  const float* adjs = (const float*)d_in[1];
  const float* conv_w = (const float*)d_in[2];
  const float* conv_b = (const float*)d_in[3];
  const float* enc_w1 = (const float*)d_in[4];
  const float* enc_w2 = (const float*)d_in[5];
  const float* enc_b = (const float*)d_in[6];
  const float* dec_w = (const float*)d_in[7];
  const float* dec_b = (const float*)d_in[8];
  const float* out_w = (const float*)d_in[9];
  const float* out_b = (const float*)d_in[10];
  float* preds = (float*)d_out;

  char* ws = (char*)d_ws;
  size_t off = 0;
  auto take = [&](size_t bytes) {
    char* p = ws + off;
    off = (off + bytes + 255) & ~(size_t)255;
    return p;
  };
  _Float16* A16 = (_Float16*)take((size_t)EE * NN * NN * 2);
  _Float16* condT16 = (_Float16*)take((size_t)BB * FF * NN * 2);
  _Float16* cond16 = (_Float16*)take((size_t)BB * NN * FF * 2);
  _Float16* agg16 = (_Float16*)take((size_t)EE * BB * NN * FF * 2);
  float* win = (float*)take((size_t)BB * NN * TT * DD * 4);
  _Float16* w1T = (_Float16*)take((size_t)EE * HH * FF * 2);
  _Float16* w2T = (_Float16*)take((size_t)EE * HH * FF * 2);
  _Float16* decT = (_Float16*)take((size_t)HH * (FF + HH) * 2);
  float* mask = (float*)take((size_t)NN * 4);

  // ---- one-time prep (constant across the 10 rollout steps) ----
  {
    int n = EE * NN * NN;
    prep_adj_kernel<<<(n + 255) / 256, 256, 0, stream>>>(adjs, A16, n);
  }
  prep_mask_kernel<<<NN / 256, 256, 0, stream>>>(adjs, mask);
  {
    int n = EE * HH * FF + HH * (FF + HH);
    prep_w_kernel<<<(n + 255) / 256, 256, 0, stream>>>(enc_w1, enc_w2, dec_w,
                                                       w1T, w2T, decT);
  }
  {
    int n = BB * NN * TT * DD;
    prep_win_kernel<<<(n + 255) / 256, 256, 0, stream>>>(time_segs, win);
  }

  // ---- rollout ----
  for (int s = 0; s < PP; ++s) {
    cond_kernel<<<(BB * NN * FF) / 256, 256, 0, stream>>>(win, conv_w, conv_b,
                                                          cond16, condT16);
    agg_gemm_kernel<<<dim3(NN / (32 * 8), EE * BB), 256, 0, stream>>>(
        A16, condT16, agg16);
    encdec_kernel<<<(BB * NN / 16) / 8, 256, 0, stream>>>(
        agg16, cond16, w1T, w2T, decT, enc_b, dec_b, out_w, out_b, mask, win,
        preds, s);
  }
}
